// MACRO_68813966016627
// MI455X (gfx1250) — matrix operand order repaired and checked by reading, not compiled, not run on hardware
//
#include <hip/hip_runtime.h>
#include <math.h>

typedef __attribute__((ext_vector_type(16))) _Float16 v16h;
typedef __attribute__((ext_vector_type(8)))  _Float16 v8h;
typedef __attribute__((ext_vector_type(8)))  float    v8f;

#define N_NODES 20000
#define N_EDGES 320000
#define DIM_IN  64
#define EMB     128
#define HID     512     // 4*EMB
#define OUT_DIM 10
#define EPSX    1e-20f

union AFrag { v16h v; v8h h[2]; _Float16 e[16]; };
union CFrag { v8f  v; float f[8]; };

// ---- ordered-float <-> uint encoding for atomic min/max ----
__device__ __forceinline__ unsigned enc_f(float f) {
    unsigned u = __float_as_uint(f);
    return (u & 0x80000000u) ? ~u : (u | 0x80000000u);
}
__device__ __forceinline__ float dec_f(unsigned e) {
    return (e & 0x80000000u) ? __uint_as_float(e ^ 0x80000000u)
                             : __uint_as_float(~e);
}

// ---------------- elementwise helpers ----------------
__global__ void k_cvt_f16(const float* __restrict__ s, _Float16* __restrict__ d, int n) {
    int i = blockIdx.x * blockDim.x + threadIdx.x;
    if (i < n) d[i] = (_Float16)s[i];
}
__global__ void k_copy(const float* __restrict__ s, float* __restrict__ d, int n) {
    int i = blockIdx.x * blockDim.x + threadIdx.x;
    if (i < n) d[i] = s[i];
}

// Swizzle a row-major fp32 weight W[K, Ncols] into WMMA B-fragment order (fp16):
// out[(((nt*(K/32)+ks)*32 + lane)*16 + i)] = W[(ks*32 + (lane>>4)*8 + (i&7) + (i>>3)*16)*Ncols + nt*16+(lane&15)]
__global__ void k_swz_w(const float* __restrict__ W, _Float16* __restrict__ out, int K, int Ncols) {
    int idx = blockIdx.x * blockDim.x + threadIdx.x;
    if (idx >= K * Ncols) return;
    int i    = idx & 15;
    int lane = (idx >> 4) & 31;
    int tile = idx >> 9;
    int KS   = K >> 5;
    int nt   = tile / KS;
    int ks   = tile - nt * KS;
    int half = lane >> 4;
    int col  = nt * 16 + (lane & 15);
    int kk   = ks * 32 + half * 8 + (i & 7) + (i >> 3) * 16;
    out[idx] = (_Float16)W[kk * Ncols + col];
}

// segment_sum: xa[dst] += h[src] * att   (xa pre-initialized with h for the GIN self term)
__global__ void k_scatter(const float* __restrict__ h, const int* __restrict__ src,
                          const int* __restrict__ dst, const float* __restrict__ att,
                          float* __restrict__ xa, int K) {
    int i = blockIdx.x * blockDim.x + threadIdx.x;
    int tot = N_EDGES * K;
    if (i >= tot) return;
    int e = i / K, k = i - e * K;
    float w = att ? att[e] : 1.0f;
    atomicAdd(&xa[dst[e] * K + k], h[src[e] * K + k] * w);
}

// ---------------- WMMA GEMM: Y[rows,128] = relu(A[rows,K] @ W[K,128] + b) ----------------
// Wswz is in fragment-swizzled order; staged once into LDS per block (shared by 8 waves).
template <int K>
__global__ __launch_bounds__(256)
void k_gemm_relu(const _Float16* __restrict__ A, const _Float16* __restrict__ Wswz,
                 const float* __restrict__ bias, float* __restrict__ Y, int rowTiles) {
    extern __shared__ _Float16 wsm[];             // K*128 halves
    const int tid  = threadIdx.x;
    const int lane = tid & 31;
    constexpr int KS = K / 32;

    // cooperative stage of the whole swizzled weight (K*128*2 bytes)
    for (int c = tid; c < K * 16; c += blockDim.x)
        ((v8h*)wsm)[c] = ((const v8h*)Wswz)[c];
    __syncthreads();

    int wid = (blockIdx.x * blockDim.x + tid) >> 5;
    if (wid >= rowTiles) return;
    const int half = lane >> 4;
    const int row  = lane & 15;
    const int m0   = wid * 16;

    AFrag a[KS];
    const int rowbase = (m0 + row) * K;
#pragma unroll
    for (int ks = 0; ks < KS; ++ks) {
        a[ks].h[0] = *(const v8h*)(A + rowbase + ks * 32 + half * 8);
        a[ks].h[1] = *(const v8h*)(A + rowbase + ks * 32 + 16 + half * 8);
    }
#pragma unroll
    for (int nt = 0; nt < 8; ++nt) {
        int col = nt * 16 + (lane & 15);
        CFrag c;
        float bv = bias[col];
#pragma unroll
        for (int r = 0; r < 8; ++r) c.f[r] = bv;
#pragma unroll
        for (int ks = 0; ks < KS; ++ks) {
            AFrag b;
            b.v = *(const v16h*)(wsm + (((nt * KS + ks) * 32 + lane) << 4));
            c.v = __builtin_amdgcn_wmma_f32_16x16x32_f16(false, a[ks].v, false, b.v,
                                                         (short)0, c.v, false, false);
        }
#pragma unroll
        for (int r = 0; r < 8; ++r) {
            float v = c.f[r];
            Y[(m0 + half * 8 + r) * EMB + col] = v > 0.f ? v : 0.f;
        }
    }
}

// ---------------- fused edge-attention MLP ----------------
// Per wave: one 16-edge tile (A = 64 VGPRs, stays resident — no spills).
// hidden = relu([h_src||h_dst] @ W1) contracted with W2 on the fly.
// W1 (swizzled, 256KB) staged through LDS in four 64KB stages, shared by all 8 waves.
__global__ __launch_bounds__(256)
void k_edge_mlp(const _Float16* __restrict__ hf, const _Float16* __restrict__ w1s,
                const float* __restrict__ b1, const float* __restrict__ w2,
                const float* __restrict__ b2, const int* __restrict__ src,
                const int* __restrict__ dst, float* __restrict__ pe) {
    extern __shared__ _Float16 bsm[];             // 64KB: 8 nt x 8 ks fragment tiles
    const int tid  = threadIdx.x;
    const int lane = tid & 31;
    const int gw   = blockIdx.x * (blockDim.x >> 5) + (tid >> 5);
    const int half = lane >> 4;
    const int row  = lane & 15;

    const int e = gw * 16 + row;
    const int s = src[e], d = dst[e];

    AFrag a[8];                                   // 16x256 fp16 A tile, 8 k-steps of 32
#pragma unroll
    for (int ks = 0; ks < 8; ++ks) {
        int kg0 = ks * 32 + half * 8;             // runs never straddle the 128 split
        int kg1 = kg0 + 16;
        const _Float16* p0 = (kg0 < EMB) ? (hf + s * EMB + kg0) : (hf + d * EMB + (kg0 - EMB));
        const _Float16* p1 = (kg1 < EMB) ? (hf + s * EMB + kg1) : (hf + d * EMB + (kg1 - EMB));
        a[ks].h[0] = *(const v8h*)p0;
        a[ks].h[1] = *(const v8h*)p1;
    }
    float acc[8];
#pragma unroll
    for (int r = 0; r < 8; ++r) acc[r] = 0.f;

    for (int st = 0; st < 4; ++st) {              // 4 stages x 8 nt tiles
        __syncthreads();                          // protect previous stage from overwrite
        for (int c = tid; c < 4096; c += 256)     // 64KB stage copy, b128 per thread-chunk
            ((v8h*)bsm)[c] = ((const v8h*)w1s)[st * 4096 + c];
        __syncthreads();
        if (st < 3)                               // prefetch part of next stage
            __builtin_prefetch(w1s + (st + 1) * 32768 + tid * 64, 0, 0);

        for (int ntl = 0; ntl < 8; ++ntl) {
            int nt  = st * 8 + ntl;
            int col = nt * 16 + (lane & 15);
            CFrag c;
            float bv = b1[col];
#pragma unroll
            for (int r = 0; r < 8; ++r) c.f[r] = bv;
#pragma unroll
            for (int ks = 0; ks < 8; ++ks) {
                AFrag b;                          // ds_load_b128 x2, conflict-free
                b.v = *(const v16h*)(bsm + (((ntl * 8 + ks) * 32 + lane) << 4));
                c.v = __builtin_amdgcn_wmma_f32_16x16x32_f16(false, a[ks].v, false, b.v,
                                                             (short)0, c.v, false, false);
            }
            float w2c = w2[col];
#pragma unroll
            for (int r = 0; r < 8; ++r) {
                float v = c.f[r];
                acc[r] += (v > 0.f ? v : 0.f) * w2c;
            }
        }
    }
    // reduce over the 16 lanes of each half; rows 0..7 live in lanes 0-15, rows 8..15 in 16-31
#pragma unroll
    for (int r = 0; r < 8; ++r) {
        float v = acc[r];
        v += __shfl_xor(v, 1, 32);
        v += __shfl_xor(v, 2, 32);
        v += __shfl_xor(v, 4, 32);
        v += __shfl_xor(v, 8, 32);
        acc[r] = v;
    }
    if ((lane & 15) == 0) {
        float bb = b2[0];
#pragma unroll
        for (int r = 0; r < 8; ++r)
            pe[gw * 16 + half * 8 + r] = acc[r] + bb;
    }
}

// ---------------- pe statistics ----------------
__global__ void k_stats_init(float* s) {
    if (threadIdx.x == 0) {
        s[0] = 0.f; s[1] = 0.f;
        ((unsigned*)s)[2] = 0u;
        ((unsigned*)s)[3] = 0u;
    }
}
__global__ void k_stats(const float* __restrict__ pe, float* s) {
    int i = blockIdx.x * blockDim.x + threadIdx.x;
    if (i >= N_EDGES) return;
    float v = pe[i];
    atomicAdd(&s[0], v);
    atomicAdd(&s[1], v * v);
    atomicMax(&((unsigned*)s)[3], enc_f(v));
    atomicMax(&((unsigned*)s)[2], enc_f(-v));
}
__global__ void k_stats_fin(float* s) {
    if (threadIdx.x != 0) return;
    float sum = s[0], sumsq = s[1];
    float mean = sum / (float)N_EDGES;
    float var  = (sumsq - sum * sum / (float)N_EDGES) / (float)(N_EDGES - 1);
    float inv  = 1.0f / sqrtf(var);
    float pmin = -dec_f(((unsigned*)s)[2]);
    float pmax =  dec_f(((unsigned*)s)[3]);
    s[4] = mean; s[5] = inv;
    s[6] = (pmin - mean) * inv;
    s[7] = (pmax - mean) * inv;
}

// ---------------- Sinkhorn in log space ----------------
__global__ void k_lt_init(const float* __restrict__ pe, const float* __restrict__ u,
                          const float* __restrict__ s, float* __restrict__ lt) {
    int i = blockIdx.x * blockDim.x + threadIdx.x;
    if (i >= N_EDGES) return;
    float atts = (pe[i] - s[4]) * s[5];
    float g = -logf(-logf(u[i] + EPSX) + EPSX);
    float d = atts + g;
    lt[2 * i]     = -(d - s[6]);
    lt[2 * i + 1] = -(s[7] - d);
}
__global__ void k_sk_reset(float* cs) {
    if (threadIdx.x == 0) {
        ((unsigned*)cs)[0] = 0u; ((unsigned*)cs)[1] = 0u;
        cs[2] = 0.f; cs[3] = 0.f;
    }
}
__global__ void k_sk_row(float* __restrict__ lt, float* cs) {
    int i = blockIdx.x * blockDim.x + threadIdx.x;
    if (i >= N_EDGES) return;
    float a = lt[2 * i], b = lt[2 * i + 1];
    float m = fmaxf(a, b);
    float lse = m + logf(expf(a - m) + expf(b - m));
    a -= lse; b -= lse;
    lt[2 * i] = a; lt[2 * i + 1] = b;
    atomicMax(&((unsigned*)cs)[0], enc_f(a));
    atomicMax(&((unsigned*)cs)[1], enc_f(b));
}
__global__ void k_sk_colsum(const float* __restrict__ lt, float* cs) {
    int i = blockIdx.x * blockDim.x + threadIdx.x;
    if (i >= N_EDGES) return;
    float m0 = dec_f(((unsigned*)cs)[0]), m1 = dec_f(((unsigned*)cs)[1]);
    atomicAdd(&cs[2], expf(lt[2 * i] - m0));
    atomicAdd(&cs[3], expf(lt[2 * i + 1] - m1));
}
__global__ void k_sk_fin(float* __restrict__ lt, const float* cs, float lr0, float lr1) {
    int i = blockIdx.x * blockDim.x + threadIdx.x;
    if (i >= N_EDGES) return;
    float m0 = dec_f(((const unsigned*)cs)[0]), m1 = dec_f(((const unsigned*)cs)[1]);
    float l0 = m0 + logf(cs[2]), l1 = m1 + logf(cs[3]);
    lt[2 * i]     = lt[2 * i]     - l0 + lr0;
    lt[2 * i + 1] = lt[2 * i + 1] - l1 + lr1;
}
__global__ void k_T(const float* __restrict__ lt, float* __restrict__ T) {
    int i = blockIdx.x * blockDim.x + threadIdx.x;
    if (i < N_EDGES) T[i] = expf(lt[2 * i + 1]);
}

// ---------------- pooling + head ----------------
__global__ void k_pool_init(float* rs) { rs[threadIdx.x] = 0.f; }
__global__ void k_pool(const float* __restrict__ hc, float* rs) {
    int i = blockIdx.x * blockDim.x + threadIdx.x;
    if (i >= N_NODES * EMB) return;
    atomicAdd(&rs[i & (EMB - 1)], hc[i]);
}
__global__ void k_head(const float* __restrict__ rs, const float* __restrict__ hW,
                       const float* __restrict__ hb, float* __restrict__ out) {
    int o = threadIdx.x;
    if (o >= OUT_DIM) return;
    float acc = hb[o];
    const float invn = 1.0f / (float)N_NODES;
    for (int k = 0; k < EMB; ++k) acc += rs[k] * invn * hW[k * OUT_DIM + o];
    out[o] = acc;
}

#define GRID(n, b) (((n) + (b) - 1) / (b))

extern "C" void kernel_launch(void* const* d_in, const int* in_sizes, int n_in,
                              void* d_out, int out_size, void* d_ws, size_t ws_size,
                              hipStream_t stream) {
    const float* x      = (const float*)d_in[0];
    const int*   ei     = (const int*)d_in[1];
    const int*   src    = ei;
    const int*   dst    = ei + N_EDGES;
    const float* u      = (const float*)d_in[2];
    const float* enc_W0 = (const float*)d_in[3];
    const float* enc_b0 = (const float*)d_in[4];
    const float* enc_W  = (const float*)d_in[5];
    const float* enc_b  = (const float*)d_in[6];
    const float* ea_W1  = (const float*)d_in[7];
    const float* ea_b1  = (const float*)d_in[8];
    const float* ea_W2  = (const float*)d_in[9];
    const float* ea_b2  = (const float*)d_in[10];
    const float* cls_W0 = (const float*)d_in[11];
    const float* cls_b0 = (const float*)d_in[12];
    const float* cls_W  = (const float*)d_in[13];
    const float* cls_b  = (const float*)d_in[14];
    const float* head_W = (const float*)d_in[15];
    const float* head_b = (const float*)d_in[16];

    // ---- carve workspace ----
    char*  base = (char*)d_ws;
    size_t off  = 0;
    auto alloc  = [&](size_t bytes) -> void* {
        void* r = base + off;
        off = (off + bytes + 255) & ~(size_t)255;
        return r;
    };
    float*     h     = (float*)alloc((size_t)N_NODES * EMB * 4);
    float*     xa    = (float*)alloc((size_t)N_NODES * EMB * 4);
    _Float16*  a16   = (_Float16*)alloc((size_t)N_NODES * EMB * 2);
    _Float16*  hf16  = (_Float16*)alloc((size_t)N_NODES * EMB * 2);
    _Float16*  ew0h  = (_Float16*)alloc((size_t)DIM_IN * EMB * 2);
    _Float16*  ewh   = (_Float16*)alloc((size_t)4 * EMB * EMB * 2);
    _Float16*  cw0h  = (_Float16*)alloc((size_t)DIM_IN * EMB * 2);
    _Float16*  cwh   = (_Float16*)alloc((size_t)4 * EMB * EMB * 2);
    _Float16*  w1h   = (_Float16*)alloc((size_t)2 * EMB * HID * 2);
    float*     pe    = (float*)alloc((size_t)N_EDGES * 4);
    float*     lt    = (float*)alloc((size_t)2 * N_EDGES * 4);
    float*     Tbuf  = (float*)alloc((size_t)N_EDGES * 4);
    float*     stat  = (float*)alloc(64);
    float*     cs    = (float*)alloc(64);
    float*     rs    = (float*)alloc(EMB * 4);

    const int B = 256;
    const int rowTiles = N_NODES / 16;                 // 1250 exact
    const int gemmGrid = GRID(rowTiles * 32, B);
    const int smW0 = DIM_IN * EMB * 2;                 // 16KB
    const int smW  = EMB * EMB * 2;                    // 32KB
    const int smE  = 65536;                            // 64KB edge-MLP stage

    // ---- fp16 weight conversion + fragment swizzle ----
    k_swz_w<<<GRID(DIM_IN * EMB, B), B, 0, stream>>>(enc_W0, ew0h, DIM_IN, EMB);
    k_swz_w<<<GRID(DIM_IN * EMB, B), B, 0, stream>>>(cls_W0, cw0h, DIM_IN, EMB);
    for (int l = 0; l < 4; ++l) {
        k_swz_w<<<GRID(EMB * EMB, B), B, 0, stream>>>(enc_W + (size_t)l * EMB * EMB,
                                                      ewh + (size_t)l * EMB * EMB, EMB, EMB);
        k_swz_w<<<GRID(EMB * EMB, B), B, 0, stream>>>(cls_W + (size_t)l * EMB * EMB,
                                                      cwh + (size_t)l * EMB * EMB, EMB, EMB);
    }
    k_swz_w<<<GRID(2 * EMB * HID, B), B, 0, stream>>>(ea_W1, w1h, 2 * EMB, HID);

    // ---- encoder GIN stack (att = 1) ----
    k_copy<<<GRID(N_NODES * DIM_IN, B), B, 0, stream>>>(x, xa, N_NODES * DIM_IN);
    k_scatter<<<GRID(N_EDGES * DIM_IN, B), B, 0, stream>>>(x, src, dst, nullptr, xa, DIM_IN);
    k_cvt_f16<<<GRID(N_NODES * DIM_IN, B), B, 0, stream>>>(xa, a16, N_NODES * DIM_IN);
    k_gemm_relu<DIM_IN><<<gemmGrid, B, smW0, stream>>>(a16, ew0h, enc_b0, h, rowTiles);
    for (int l = 0; l < 4; ++l) {
        k_copy<<<GRID(N_NODES * EMB, B), B, 0, stream>>>(h, xa, N_NODES * EMB);
        k_scatter<<<GRID(N_EDGES * EMB, B), B, 0, stream>>>(h, src, dst, nullptr, xa, EMB);
        k_cvt_f16<<<GRID(N_NODES * EMB, B), B, 0, stream>>>(xa, a16, N_NODES * EMB);
        k_gemm_relu<EMB><<<gemmGrid, B, smW, stream>>>(a16, ewh + (size_t)l * EMB * EMB,
                                                       enc_b + l * EMB, h, rowTiles);
    }
    k_cvt_f16<<<GRID(N_NODES * EMB, B), B, 0, stream>>>(h, hf16, N_NODES * EMB);

    // ---- fused edge-attention MLP (dominant WMMA work) ----
    // 20000 edge tiles / 8 waves per block = 2500 blocks (exact)
    k_edge_mlp<<<N_EDGES / 16 / 8, B, smE, stream>>>(hf16, w1h, ea_b1, ea_W2, ea_b2,
                                                     src, dst, pe);

    // ---- pe stats -> atts normalization params ----
    k_stats_init<<<1, 32, 0, stream>>>(stat);
    k_stats<<<GRID(N_EDGES, B), B, 0, stream>>>(pe, stat);
    k_stats_fin<<<1, 32, 0, stream>>>(stat);

    // ---- Sinkhorn ----
    k_lt_init<<<GRID(N_EDGES, B), B, 0, stream>>>(pe, u, stat, lt);
    const float lr0 = logf((float)N_EDGES * 0.75f + EPSX);
    const float lr1 = logf((float)N_EDGES * 0.25f + EPSX);
    for (int it = 0; it < 30; ++it) {
        k_sk_reset<<<1, 32, 0, stream>>>(cs);
        k_sk_row<<<GRID(N_EDGES, B), B, 0, stream>>>(lt, cs);
        k_sk_colsum<<<GRID(N_EDGES, B), B, 0, stream>>>(lt, cs);
        k_sk_fin<<<GRID(N_EDGES, B), B, 0, stream>>>(lt, cs, lr0, lr1);
    }
    k_T<<<GRID(N_EDGES, B), B, 0, stream>>>(lt, Tbuf);

    // ---- classifier GIN stack (att = T) ----
    k_copy<<<GRID(N_NODES * DIM_IN, B), B, 0, stream>>>(x, xa, N_NODES * DIM_IN);
    k_scatter<<<GRID(N_EDGES * DIM_IN, B), B, 0, stream>>>(x, src, dst, Tbuf, xa, DIM_IN);
    k_cvt_f16<<<GRID(N_NODES * DIM_IN, B), B, 0, stream>>>(xa, a16, N_NODES * DIM_IN);
    k_gemm_relu<DIM_IN><<<gemmGrid, B, smW0, stream>>>(a16, cw0h, cls_b0, h, rowTiles);
    for (int l = 0; l < 4; ++l) {
        k_copy<<<GRID(N_NODES * EMB, B), B, 0, stream>>>(h, xa, N_NODES * EMB);
        k_scatter<<<GRID(N_EDGES * EMB, B), B, 0, stream>>>(h, src, dst, Tbuf, xa, EMB);
        k_cvt_f16<<<GRID(N_NODES * EMB, B), B, 0, stream>>>(xa, a16, N_NODES * EMB);
        k_gemm_relu<EMB><<<gemmGrid, B, smW, stream>>>(a16, cwh + (size_t)l * EMB * EMB,
                                                       cls_b + l * EMB, h, rowTiles);
    }

    // ---- mean pool + head ----
    k_pool_init<<<1, EMB, 0, stream>>>(rs);
    k_pool<<<GRID(N_NODES * EMB, B), B, 0, stream>>>(h, rs);
    k_head<<<1, 32, 0, stream>>>(rs, head_W, head_b, (float*)d_out);
}
